// SimplifiedDualRouter_83846351553129
// MI455X (gfx1250) — matrix operand order repaired and checked by reading, not compiled, not run on hardware
//
#include <hip/hip_runtime.h>
#include <hip/hip_bf16.h>
#include <math.h>

typedef __attribute__((ext_vector_type(16))) __bf16 v16bf;
typedef __attribute__((ext_vector_type(8)))  __bf16 v8bf;
typedef __attribute__((ext_vector_type(8)))  float  v8f;

#define LN_EPS_F 1e-5f

__device__ __forceinline__ v16bf cat16(v8bf lo, v8bf hi) {
    return __builtin_shufflevector(lo, hi, 0,1,2,3,4,5,6,7,8,9,10,11,12,13,14,15);
}
__device__ __forceinline__ float gelu_f(float x) {
    return 0.5f * x * (1.0f + erff(x * 0.70710678118654752f));
}
__device__ __forceinline__ float wred_sum(float v) {
#pragma unroll
    for (int o = 16; o > 0; o >>= 1) v += __shfl_xor(v, o, 32);
    return v;
}
__device__ __forceinline__ float wred_max(float v) {
#pragma unroll
    for (int o = 16; o > 0; o >>= 1) v = fmaxf(v, __shfl_xor(v, o, 32));
    return v;
}
// 256-thread (8-wave) block reductions through 8-slot LDS
__device__ __forceinline__ float bred_sum(float v, float* red, int tid) {
    v = wred_sum(v);
    if ((tid & 31) == 0) red[tid >> 5] = v;
    __syncthreads();
    float r = red[0]+red[1]+red[2]+red[3]+red[4]+red[5]+red[6]+red[7];
    __syncthreads();
    return r;
}
__device__ __forceinline__ float bred_max(float v, float* red, int tid) {
    v = wred_max(v);
    if ((tid & 31) == 0) red[tid >> 5] = v;
    __syncthreads();
    float r = fmaxf(fmaxf(fmaxf(red[0],red[1]),fmaxf(red[2],red[3])),
                    fmaxf(fmaxf(red[4],red[5]),fmaxf(red[6],red[7])));
    __syncthreads();
    return r;
}

// ---------------------------------------------------------------------------
// Prep kernel: pre0[e][g] = expert_emb[e] . gnn_w0_top[:,g]   (256x128, K=128)
// ---------------------------------------------------------------------------
__global__ __launch_bounds__(128) void pre0_kernel(const float* __restrict__ emb,
                                                   const float* __restrict__ w0,
                                                   float* __restrict__ pre0) {
    const int e = blockIdx.x;
    const int g = threadIdx.x;
    float s = 0.0f;
#pragma unroll 8
    for (int k = 0; k < 128; ++k) s = fmaf(emb[e * 128 + k], w0[k * 128 + g], s);
    pre0[e * 128 + g] = s;
}

// Prep kernel: bf16 copy of gnn_w1 top half (128x128, row-major [k][g])
__global__ __launch_bounds__(256) void convw1_kernel(const float* __restrict__ w1,
                                                     __bf16* __restrict__ w1tb) {
    int i = blockIdx.x * 256 + threadIdx.x;
    if (i < 128 * 128) w1tb[i] = (__bf16)w1[i];
}

// ---------------------------------------------------------------------------
// Tiled bf16 WMMA GEMM: C[M,N] = epilogue(A[M,K] @ B[K,N] + bias[N])
// 128x64 block tile, BK=32, 256 threads = 8 waves, 32x32 output per wave:
// per k-step each wave does 2 A-frags + 2 B-frags (8 x ds_load_b128) feeding
// 4 WMMAs (2 LDS loads per WMMA, each fragment reused twice).
// A is f32 (converted while staging) or already bf16; B is f32 (converted).
// ---------------------------------------------------------------------------
template <bool A_BF16, bool DO_GELU, bool OUT_BF16>
__global__ __launch_bounds__(256) void gemm_wmma(const void* __restrict__ Av,
                                                 const float* __restrict__ B,
                                                 const float* __restrict__ bias,
                                                 void* __restrict__ Cv,
                                                 int M, int N, int K) {
    __shared__ __bf16 As[128][40];  // [m][k], padded row stride 80B
    __shared__ __bf16 Bst[64][40];  // transposed: [n][k], padded

    const int tid  = threadIdx.x;
    const int wave = tid >> 5;
    const int lane = tid & 31;
    const int l    = lane & 15;
    const int hi   = lane >> 4;
    const int m0   = blockIdx.y * 128;
    const int n0   = blockIdx.x * 64;
    const int wm   = wave & 3;   // 32-row wave tile
    const int wn   = wave >> 2;  // 32-col wave tile

    const int sarow = tid >> 1, sacol = (tid & 1) * 16;  // A staging: 128x32
    const int sbrow = tid >> 3, sbcol = (tid & 7) * 8;   // B staging: 32x64

    v8f acc[2][2] = {};

    for (int k0 = 0; k0 < K; k0 += 32) {
        __syncthreads();
        if (A_BF16) {
            const __bf16* A = (const __bf16*)Av;
            const __bf16* ap = &A[(size_t)(m0 + sarow) * K + k0 + sacol];
            *(v8bf*)&As[sarow][sacol]     = *(const v8bf*)ap;
            *(v8bf*)&As[sarow][sacol + 8] = *(const v8bf*)(ap + 8);
        } else {
            const float* A  = (const float*)Av;
            const float* ap = &A[(size_t)(m0 + sarow) * K + k0 + sacol];
#pragma unroll
            for (int i = 0; i < 16; ++i) As[sarow][sacol + i] = (__bf16)ap[i];
        }
        {
            const float* bp = &B[(size_t)(k0 + sbrow) * N + n0 + sbcol];
#pragma unroll
            for (int i = 0; i < 8; ++i) Bst[sbcol + i][sbrow] = (__bf16)bp[i];
        }
        __syncthreads();

        // A fragments (16x32 bf16): lane<16 -> K 0-7 & 16-23; lane>=16 -> K 8-15 & 24-31
        const int r0 = wm * 32 + l;
        v16bf a0 = cat16(*(const v8bf*)&As[r0][hi * 8],
                         *(const v8bf*)&As[r0][16 + hi * 8]);
        v16bf a1 = cat16(*(const v8bf*)&As[r0 + 16][hi * 8],
                         *(const v8bf*)&As[r0 + 16][16 + hi * 8]);
        // B fragments (32x16): lane<16 -> K 0-7 & 16-23 of col; lane>=16 -> K 8-15 & 24-31
        const int cr0 = wn * 32 + l;
        v16bf b0 = cat16(*(const v8bf*)&Bst[cr0][hi * 8],
                         *(const v8bf*)&Bst[cr0][16 + hi * 8]);
        v16bf b1 = cat16(*(const v8bf*)&Bst[cr0 + 16][hi * 8],
                         *(const v8bf*)&Bst[cr0 + 16][16 + hi * 8]);
        acc[0][0] = __builtin_amdgcn_wmma_f32_16x16x32_bf16(false, a0, false, b0,
                                                            (short)0, acc[0][0], false, false);
        acc[0][1] = __builtin_amdgcn_wmma_f32_16x16x32_bf16(false, a0, false, b1,
                                                            (short)0, acc[0][1], false, false);
        acc[1][0] = __builtin_amdgcn_wmma_f32_16x16x32_bf16(false, a1, false, b0,
                                                            (short)0, acc[1][0], false, false);
        acc[1][1] = __builtin_amdgcn_wmma_f32_16x16x32_bf16(false, a1, false, b1,
                                                            (short)0, acc[1][1], false, false);
    }

    const int mh = hi * 8;
#pragma unroll
    for (int ti = 0; ti < 2; ++ti) {
#pragma unroll
        for (int tj = 0; tj < 2; ++tj) {
            const int col = n0 + wn * 32 + tj * 16 + l;
            const float bv = bias[col];
#pragma unroll
            for (int r = 0; r < 8; ++r) {
                float v = acc[ti][tj][r] + bv;
                if (DO_GELU) v = gelu_f(v);
                const int row = m0 + wm * 32 + ti * 16 + mh + r;
                if (OUT_BF16) ((__bf16*)Cv)[(size_t)row * N + col] = (__bf16)v;
                else          ((float*)Cv)[(size_t)row * N + col]  = v;
            }
        }
    }
}

// ---------------------------------------------------------------------------
// Fused per-row router: one block (8 waves) per row n.
// fast softmax -> confidence; GNN layer0 (closed form) -> node;
// layer1 node@w1_top via WMMA (64-expert chunks); scores; blend.
// ---------------------------------------------------------------------------
__global__ __launch_bounds__(256) void router_kernel(
    const float* __restrict__ fast_logits, const float* __restrict__ query,
    const float* __restrict__ pre0, const __bf16* __restrict__ w1tb,
    const float* __restrict__ expert_emb,
    const float* __restrict__ gnn_w0, const float* __restrict__ gnn_b0,
    const float* __restrict__ ln_g0, const float* __restrict__ ln_b0,
    const float* __restrict__ gnn_w1, const float* __restrict__ gnn_b1,
    const float* __restrict__ ln_g1, const float* __restrict__ ln_b1,
    const float* __restrict__ out_w, const float* __restrict__ out_b,
    float* __restrict__ out) {
    extern __shared__ char dyn[];
    __bf16* w1tT  = (__bf16*)dyn;                               // [128][136] bf16 (w1_top^T)
    __bf16* nodeb = (__bf16*)(dyn + 128 * 136 * 2);             // [64][136] bf16
    float*  nodef = (float*)(dyn + (128 + 64) * 136 * 2);       // [64][128] f32
    float*  mm    = nodef + 64 * 128;                           // [64][128] f32

    __shared__ float fastp[256];
    __shared__ float scores[256];
    __shared__ float qv[128];
    __shared__ float qp0[128];
    __shared__ float qp1[128];
    __shared__ float red[8];

    const int n    = blockIdx.x;
    const int t    = threadIdx.x;
    const int wave = t >> 5;
    const int lane = t & 31;
    const int l    = lane & 15;
    const int hi   = lane >> 4;

    // ---- fast pathway softmax; confidence = max prob = 1/sum(exp(v-max)) ----
    float lg   = fast_logits[(size_t)n * 256 + t];
    float fmx  = bred_max(lg, red, t);
    float fe   = expf(lg - fmx);
    float fsum = bred_sum(fe, red, t);
    fastp[t]   = fe;
    const float conf = 1.0f / fsum;

    // ---- query row + query @ w{0,1}_bot + b ----
    if (t < 128) qv[t] = query[(size_t)n * 128 + t];
    __syncthreads();
    {
        const int g = t & 127;
        const float* wmat = (t < 128) ? gnn_w0 : gnn_w1;
        float s = (t < 128) ? gnn_b0[g] : gnn_b1[g];
        for (int k = 0; k < 128; ++k) s = fmaf(qv[k], wmat[(128 + k) * 128 + g], s);
        if (t < 128) qp0[g] = s; else qp1[g] = s;
    }
    // stage w1_top^T into LDS (reused by every k-step / chunk)
    for (int i = t; i < 128 * 128; i += 256)
        w1tT[(i & 127) * 136 + (i >> 7)] = w1tb[i];
    __syncthreads();

    const int tm  = wave & 3;          // M tile (16 experts)
    const int tnb = (wave >> 2) * 4;   // first of 4 N tiles

    for (int ch = 0; ch < 4; ++ch) {
        const int ebase = ch * 64;

        // -- layer 0: wave-per-expert, lane holds 4 g-values --
        for (int pass = 0; pass < 8; ++pass) {
            const int el = pass * 8 + wave;
            const int e  = ebase + el;
            const int g  = lane * 4;
            float val[4];
#pragma unroll
            for (int j = 0; j < 4; ++j)
                val[j] = gelu_f(pre0[e * 128 + g + j] + qp0[g + j]);
            float s  = val[0] + val[1] + val[2] + val[3];
            float ss = val[0]*val[0] + val[1]*val[1] + val[2]*val[2] + val[3]*val[3];
            s  = wred_sum(s);
            ss = wred_sum(ss);
            const float mean = s * 0.0078125f;
            const float var  = ss * 0.0078125f - mean * mean;
            const float inv  = rsqrtf(var + LN_EPS_F);
#pragma unroll
            for (int j = 0; j < 4; ++j) {
                float u  = (val[j] - mean) * inv * ln_g0[g + j] + ln_b0[g + j];
                float nv = expert_emb[(size_t)e * 128 + g + j] + u;
                nodef[el * 128 + g + j] = nv;
                nodeb[el * 136 + g + j] = (__bf16)nv;
            }
        }
        __syncthreads();

        // -- layer 1 matmul: mm[64][128] = nodeb @ w1_top (+qp1) via WMMA --
        {
            v8f acc[4] = {};
#pragma unroll
            for (int kk = 0; kk < 4; ++kk) {
                const int kb = kk * 32;
                const __bf16* ap = &nodeb[(tm * 16 + l) * 136 + kb + hi * 8];
                v16bf af = cat16(*(const v8bf*)ap, *(const v8bf*)(ap + 16));
#pragma unroll
                for (int j = 0; j < 4; ++j) {
                    const __bf16* bp = &w1tT[((tnb + j) * 16 + l) * 136 + kb + hi * 8];
                    v16bf bf = cat16(*(const v8bf*)bp, *(const v8bf*)(bp + 16));
                    acc[j] = __builtin_amdgcn_wmma_f32_16x16x32_bf16(
                        false, af, false, bf, (short)0, acc[j], false, false);
                }
            }
            const int mh = hi * 8;
#pragma unroll
            for (int j = 0; j < 4; ++j) {
                const int col = (tnb + j) * 16 + l;
                const float qb = qp1[col];
#pragma unroll
                for (int r = 0; r < 8; ++r)
                    mm[(tm * 16 + mh + r) * 128 + col] = acc[j][r] + qb;
            }
        }
        __syncthreads();

        // -- layer 1 GELU+LN+residual + score dot --
        for (int pass = 0; pass < 8; ++pass) {
            const int el = pass * 8 + wave;
            const int e  = ebase + el;
            const int g  = lane * 4;
            float val[4];
#pragma unroll
            for (int j = 0; j < 4; ++j) val[j] = gelu_f(mm[el * 128 + g + j]);
            float s  = val[0] + val[1] + val[2] + val[3];
            float ss = val[0]*val[0] + val[1]*val[1] + val[2]*val[2] + val[3]*val[3];
            s  = wred_sum(s);
            ss = wred_sum(ss);
            const float mean = s * 0.0078125f;
            const float var  = ss * 0.0078125f - mean * mean;
            const float inv  = rsqrtf(var + LN_EPS_F);
            float sc = 0.0f;
#pragma unroll
            for (int j = 0; j < 4; ++j) {
                float u  = (val[j] - mean) * inv * ln_g1[g + j] + ln_b1[g + j];
                float nv = nodef[el * 128 + g + j] + u;
                sc = fmaf(nv, out_w[g + j], sc);
            }
            sc = wred_sum(sc);
            if (lane == 0) scores[e] = sc + out_b[0];
        }
        __syncthreads();
    }

    // ---- slow softmax + sigmoid blend + renormalize ----
    float scv  = scores[t];
    float smx  = bred_max(scv, red, t);
    float se   = expf(scv - smx);
    float ssum = bred_sum(se, red, t);
    float bw   = 1.0f / (1.0f + expf(-10.0f * (conf - 0.75f)));
    float fin  = bw * (fastp[t] / fsum) + (1.0f - bw) * (se / ssum);
    float tot  = bred_sum(fin, red, t);
    out[(size_t)n * 256 + t] = fin / (tot + 1e-8f);
}

// ---------------------------------------------------------------------------
extern "C" void kernel_launch(void* const* d_in, const int* in_sizes, int n_in,
                              void* d_out, int out_size, void* d_ws, size_t ws_size,
                              hipStream_t stream) {
    (void)in_sizes; (void)n_in; (void)out_size; (void)ws_size;
    const float* x      = (const float*)d_in[0];
    const float* fg_w1  = (const float*)d_in[1];
    const float* fg_b1  = (const float*)d_in[2];
    const float* fg_w2  = (const float*)d_in[3];
    const float* fg_b2  = (const float*)d_in[4];
    const float* q_w    = (const float*)d_in[5];
    const float* q_b    = (const float*)d_in[6];
    const float* emb    = (const float*)d_in[7];
    const float* gnn_w0 = (const float*)d_in[8];
    const float* gnn_b0 = (const float*)d_in[9];
    const float* ln_g0  = (const float*)d_in[10];
    const float* ln_b0  = (const float*)d_in[11];
    const float* gnn_w1 = (const float*)d_in[12];
    const float* gnn_b1 = (const float*)d_in[13];
    const float* ln_g1  = (const float*)d_in[14];
    const float* ln_b1  = (const float*)d_in[15];
    const float* out_w  = (const float*)d_in[16];
    const float* out_b  = (const float*)d_in[17];
    float* out = (float*)d_out;

    const int N = 4096, H = 5120, Hh = 2560, E = 256, G = 128;

    char* ws = (char*)d_ws;
    __bf16* h_bf  = (__bf16*)ws;  ws += (size_t)N * Hh * 2;   // 20.97 MB
    float*  fastl = (float*)ws;   ws += (size_t)N * E * 4;    //  4.19 MB
    float*  qry   = (float*)ws;   ws += (size_t)N * G * 4;    //  2.10 MB
    float*  pre0  = (float*)ws;   ws += (size_t)E * G * 4;    //  0.13 MB
    __bf16* w1tb  = (__bf16*)ws;  ws += (size_t)G * G * 2;    //  0.03 MB

    pre0_kernel<<<dim3(E), dim3(G), 0, stream>>>(emb, gnn_w0, pre0);
    convw1_kernel<<<dim3(64), dim3(256), 0, stream>>>(gnn_w1, w1tb);

    // h = gelu(x @ fg_w1 + b1)            [4096, 2560] bf16
    gemm_wmma<false, true, true><<<dim3(Hh / 64, N / 128), dim3(256), 0, stream>>>(
        x, fg_w1, fg_b1, h_bf, N, Hh, H);
    // fast_logits = h @ fg_w2 + b2        [4096, 256] f32
    gemm_wmma<true, false, false><<<dim3(E / 64, N / 128), dim3(256), 0, stream>>>(
        h_bf, fg_w2, fg_b2, fastl, N, E, Hh);
    // query = x @ q_w + q_b               [4096, 128] f32
    gemm_wmma<false, false, false><<<dim3(G / 64, N / 128), dim3(256), 0, stream>>>(
        x, q_w, q_b, qry, N, G, H);

    const size_t dynBytes = (size_t)(128 * 136 * 2)   // w1_top^T bf16
                          + (size_t)(64 * 136 * 2)    // node bf16
                          + (size_t)(64 * 128 * 4)    // node f32
                          + (size_t)(64 * 128 * 4);   // mm f32   -> 117760 B
    (void)hipFuncSetAttribute(reinterpret_cast<const void*>(router_kernel),
                              hipFuncAttributeMaxDynamicSharedMemorySize,
                              (int)dynBytes);
    router_kernel<<<dim3(N), dim3(256), dynBytes, stream>>>(
        fastl, qry, pre0, w1tb, emb, gnn_w0, gnn_b0, ln_g0, ln_b0,
        gnn_w1, gnn_b1, ln_g1, ln_b1, out_w, out_b, out);
}
